// PTDNetGCN_54245436949038
// MI455X (gfx1250) — compile-verified, matrix operand order checked by reading, not run on hardware
//
#include <hip/hip_runtime.h>
#include <hip/hip_bf16.h>

typedef __attribute__((ext_vector_type(16))) __bf16 v16bf;
typedef __attribute__((ext_vector_type(8)))  float  v8f;

#define IN_DIM 512
#define H0 32
#define H1 8
#define DH 8

// ---------------------------------------------------------------------------
// Pack W0 (512x32, row-major) into WMMA B-fragment layout, bf16.
// Fragment element index: ((sect*16 + kiter)*32 + lane)*16 + i
//   sect  = which 16-col group, kiter = k0/32
//   lane  : N = sect*16 + (lane&15); K-half = (lane<16)?0:8
//   i     : K = kiter*32 + (i&7) + ((i<8)?0:16) + khalf
// ---------------------------------------------------------------------------
__global__ void ptdnet_pack_b1(const float* __restrict__ W0, __bf16* __restrict__ Bp) {
    int idx = blockIdx.x * blockDim.x + threadIdx.x;   // 2*16*32*16 = 16384
    if (idx >= 16384) return;
    int i     = idx & 15;
    int lane  = (idx >> 4) & 31;
    int kiter = (idx >> 9) & 15;
    int sect  = idx >> 13;
    int K = kiter * 32 + (i & 7) + ((i < 8) ? 0 : 16) + ((lane < 16) ? 0 : 8);
    int N = sect * 16 + (lane & 15);
    Bp[idx] = (__bf16)W0[K * H0 + N];
}

// Pack W1 (32x8) into one 32x16 B fragment (cols 8..15 zero), bf16.
__global__ void ptdnet_pack_b2(const float* __restrict__ W1, __bf16* __restrict__ Bp) {
    int idx = blockIdx.x * blockDim.x + threadIdx.x;   // 32*16 = 512
    if (idx >= 512) return;
    int i    = idx & 15;
    int lane = idx >> 4;
    int K = (i & 7) + ((i < 8) ? 0 : 16) + ((lane < 16) ? 0 : 8);
    int N = lane & 15;
    Bp[idx] = (N < H1) ? (__bf16)W1[K * H1 + N] : (__bf16)0.0f;
}

// Build a 16-element bf16 A-fragment from a row pointer at (row, k0+khalf):
// elements 0..7  <- p[0..7]   (K = khalf + 0..7)
// elements 8..15 <- p[16..23] (K = khalf + 16..23)
__device__ inline v16bf ptdnet_load_a_frag(const float* __restrict__ p) {
    float4 a0 = *(const float4*)(p);
    float4 a1 = *(const float4*)(p + 4);
    float4 a2 = *(const float4*)(p + 16);
    float4 a3 = *(const float4*)(p + 20);
    v16bf a;
    a[0]  = (__bf16)a0.x; a[1]  = (__bf16)a0.y; a[2]  = (__bf16)a0.z; a[3]  = (__bf16)a0.w;
    a[4]  = (__bf16)a1.x; a[5]  = (__bf16)a1.y; a[6]  = (__bf16)a1.z; a[7]  = (__bf16)a1.w;
    a[8]  = (__bf16)a2.x; a[9]  = (__bf16)a2.y; a[10] = (__bf16)a2.z; a[11] = (__bf16)a2.w;
    a[12] = (__bf16)a3.x; a[13] = (__bf16)a3.y; a[14] = (__bf16)a3.z; a[15] = (__bf16)a3.w;
    return a;
}

// ---------------------------------------------------------------------------
// GEMM1: xw[N,32] = x[N,512] @ W0[512,32].
// One wave per 16-row tile; A fragment loaded ONCE and fed to two
// independent WMMAs (output col groups 0..15 and 16..31) -> halves the
// x-side load/convert work vs. a 2-wave version.
// ---------------------------------------------------------------------------
__global__ void ptdnet_gemm_xw(const float* __restrict__ x,
                               const __bf16* __restrict__ Bp,
                               float* __restrict__ xw, int nNodes) {
    int lane = threadIdx.x & 31;
    int tile = blockIdx.x;
    int M    = lane & 15;
    int row  = tile * 16 + M;
    if (row >= nNodes) row = nNodes - 1;  // clamp loads; stores guarded below
    int khalf = (lane < 16) ? 0 : 8;

    const float*  xrow  = x + (size_t)row * IN_DIM + khalf;
    const __bf16* bptr0 = Bp + ((size_t)lane) * 16;                  // sect 0
    const __bf16* bptr1 = Bp + ((size_t)(16 * 32) + lane) * 16;      // sect 1

    v8f c0 = {};
    v8f c1 = {};
#pragma unroll 4
    for (int kk = 0; kk < IN_DIM / 32; ++kk) {
        const float* p = xrow + kk * 32;
        __builtin_prefetch(p + 32, 0, 3);                 // global_prefetch_b8
        v16bf a  = ptdnet_load_a_frag(p);
        v16bf b0 = *(const v16bf*)(bptr0 + (size_t)kk * 32 * 16);
        v16bf b1 = *(const v16bf*)(bptr1 + (size_t)kk * 32 * 16);
        c0 = __builtin_amdgcn_wmma_f32_16x16x32_bf16(false, a, false, b0,
                                                     (short)0, c0, false, false);
        c1 = __builtin_amdgcn_wmma_f32_16x16x32_bf16(false, a, false, b1,
                                                     (short)0, c1, false, false);
    }

    int col0  = lane & 15;
    int rbase = tile * 16 + ((lane < 16) ? 0 : 8);
#pragma unroll
    for (int v = 0; v < 8; ++v) {
        int r = rbase + v;
        if (r < nNodes) {
            xw[(size_t)r * H0 + col0]      = c0[v];
            xw[(size_t)r * H0 + 16 + col0] = c1[v];
        }
    }
}

// ---------------------------------------------------------------------------
// GEMM2: xw2[N,8] = h[N,32] @ W1[32,8]; 1 wave/block, K=32 -> single WMMA.
// ---------------------------------------------------------------------------
__global__ void ptdnet_gemm_xw2(const float* __restrict__ h,
                                const __bf16* __restrict__ Bp2,
                                float* __restrict__ xw2, int nNodes) {
    int lane = threadIdx.x & 31;
    int tile = blockIdx.x;
    int M    = lane & 15;
    int row  = tile * 16 + M;
    if (row >= nNodes) row = nNodes - 1;
    int khalf = (lane < 16) ? 0 : 8;

    v16bf a = ptdnet_load_a_frag(h + (size_t)row * H0 + khalf);
    v16bf b = *(const v16bf*)(Bp2 + (size_t)lane * 16);
    v8f c = {};
    c = __builtin_amdgcn_wmma_f32_16x16x32_bf16(false, a, false, b,
                                                (short)0, c, false, false);

    int col = lane & 15;
    if (col < H1) {
        int rbase = tile * 16 + ((lane < 16) ? 0 : 8);
#pragma unroll
        for (int v = 0; v < 8; ++v) {
            int r = rbase + v;
            if (r < nNodes) xw2[(size_t)r * H1 + col] = c[v];
        }
    }
}

// ---------------------------------------------------------------------------
// Aggregation helpers
// ---------------------------------------------------------------------------
__global__ void ptdnet_degree_unit(const int* __restrict__ col,
                                   float* __restrict__ deg, int E) {
    int e = blockIdx.x * blockDim.x + threadIdx.x;
    if (e < E) atomicAdd(&deg[col[e]], 1.0f);
}

__global__ void ptdnet_degree_w(const int* __restrict__ col,
                                const float* __restrict__ mw,
                                float* __restrict__ deg, int E) {
    int e = blockIdx.x * blockDim.x + threadIdx.x;
    if (e < E) atomicAdd(&deg[col[e]], mw[e]);
}

// dinv = rsqrt(deg_edges + 1)   (+1 = self-loop weight; always > 0)
__global__ void ptdnet_rsqrt(const float* __restrict__ deg,
                             float* __restrict__ dinv, int n) {
    int i = blockIdx.x * blockDim.x + threadIdx.x;
    if (i < n) dinv[i] = rsqrtf(deg[i] + 1.0f);
}

// acc1[c] += xw[r] * dinv[r]*dinv[c]   (32-wide rows)
__global__ void ptdnet_scatter1(const float* __restrict__ xw,
                                const int* __restrict__ row,
                                const int* __restrict__ col,
                                const float* __restrict__ dinv,
                                float* __restrict__ acc, int E) {
    int e = blockIdx.x * blockDim.x + threadIdx.x;
    if (e >= E) return;
    int r = row[e], c = col[e];
    float s = dinv[r] * dinv[c];
    const float4* src = (const float4*)(xw + (size_t)r * H0);
    float* dst = acc + (size_t)c * H0;
#pragma unroll
    for (int q = 0; q < H0 / 4; ++q) {
        float4 v = src[q];
        atomicAdd(dst + q * 4 + 0, v.x * s);
        atomicAdd(dst + q * 4 + 1, v.y * s);
        atomicAdd(dst + q * 4 + 2, v.z * s);
        atomicAdd(dst + q * 4 + 3, v.w * s);
    }
}

// h = acc1 + xw*dinv^2 (self loop) + b0
__global__ void ptdnet_finalize_h(const float* __restrict__ acc,
                                  const float* __restrict__ xw,
                                  const float* __restrict__ dinv,
                                  const float* __restrict__ b0,
                                  float* __restrict__ h, int nNodes) {
    int idx = blockIdx.x * blockDim.x + threadIdx.x;
    if (idx >= nNodes * H0) return;
    int i = idx / H0, f = idx % H0;
    float d = dinv[i];
    h[idx] = acc[idx] + xw[idx] * d * d + b0[f];
}

// Per-edge attention MLP -> masked edge weight.
__global__ void ptdnet_attention(const float* __restrict__ h,
                                 const int* __restrict__ row,
                                 const int* __restrict__ col,
                                 const float* __restrict__ Wnb,
                                 const float* __restrict__ bnb,
                                 const float* __restrict__ Wself,
                                 const float* __restrict__ bself,
                                 const float* __restrict__ Watt,
                                 const float* __restrict__ batt,
                                 float* __restrict__ mw, int E) {
    int e = blockIdx.x * blockDim.x + threadIdx.x;
    if (e >= E) return;
    int r = row[e], c = col[e];
    const float* hr = h + (size_t)r * H0;
    const float* hc = h + (size_t)c * H0;
    float f1[DH], f2[DH];
#pragma unroll
    for (int j = 0; j < DH; ++j) { f1[j] = bnb[j]; f2[j] = bself[j]; }
#pragma unroll 4
    for (int k = 0; k < H0; ++k) {
        float a = hr[k], b = hc[k];
#pragma unroll
        for (int j = 0; j < DH; ++j) {
            f1[j] = fmaf(a, Wnb[k * DH + j], f1[j]);
            f2[j] = fmaf(b, Wself[k * DH + j], f2[j]);
        }
    }
    float w = batt[0];
#pragma unroll
    for (int j = 0; j < DH; ++j) w += f1[j] * Watt[j] + f2[j] * Watt[DH + j];
    w = fmaxf(w, 0.0f);                                    // relu
    float sig  = 1.0f / (1.0f + __expf(-w));               // sigmoid
    float mask = fminf(fmaxf(sig * 1.01f, 0.0f), 1.0f);    // zeta=1.01, gamma=0
    mw[e] = mask * w;
}

// out[c] += xw2[r] * dinv2[r]*mw*dinv2[c]   (8-wide rows)
__global__ void ptdnet_scatter2(const float* __restrict__ xw2,
                                const int* __restrict__ row,
                                const int* __restrict__ col,
                                const float* __restrict__ dinv,
                                const float* __restrict__ mw,
                                float* __restrict__ out, int E) {
    int e = blockIdx.x * blockDim.x + threadIdx.x;
    if (e >= E) return;
    int r = row[e], c = col[e];
    float s = dinv[r] * mw[e] * dinv[c];
    const float4* src = (const float4*)(xw2 + (size_t)r * H1);
    float* dst = out + (size_t)c * H1;
#pragma unroll
    for (int q = 0; q < H1 / 4; ++q) {
        float4 v = src[q];
        atomicAdd(dst + q * 4 + 0, v.x * s);
        atomicAdd(dst + q * 4 + 1, v.y * s);
        atomicAdd(dst + q * 4 + 2, v.z * s);
        atomicAdd(dst + q * 4 + 3, v.w * s);
    }
}

// out += xw2*dinv2^2 (self loop, weight 1) + b1
__global__ void ptdnet_finalize_out(const float* __restrict__ xw2,
                                    const float* __restrict__ dinv,
                                    const float* __restrict__ b1,
                                    float* __restrict__ out, int nNodes) {
    int idx = blockIdx.x * blockDim.x + threadIdx.x;
    if (idx >= nNodes * H1) return;
    int i = idx / H1, f = idx % H1;
    float d = dinv[i];
    out[idx] += xw2[idx] * d * d + b1[f];
}

// ---------------------------------------------------------------------------
extern "C" void kernel_launch(void* const* d_in, const int* in_sizes, int n_in,
                              void* d_out, int out_size, void* d_ws, size_t ws_size,
                              hipStream_t stream) {
    const float* x     = (const float*)d_in[0];
    const int*   eidx  = (const int*)d_in[1];
    const float* W0    = (const float*)d_in[2];
    const float* b0    = (const float*)d_in[3];
    const float* W1    = (const float*)d_in[4];
    const float* b1    = (const float*)d_in[5];
    const float* Wnb   = (const float*)d_in[6];
    const float* bnb   = (const float*)d_in[7];
    const float* Wself = (const float*)d_in[8];
    const float* bself = (const float*)d_in[9];
    const float* Watt  = (const float*)d_in[10];
    const float* batt  = (const float*)d_in[11];
    float* out = (float*)d_out;

    const int N = in_sizes[0] / IN_DIM;   // 100000
    const int E = in_sizes[1] / 2;        // 1600000
    const int* row = eidx;
    const int* col = eidx + E;

    // Workspace layout (256B aligned slices)
    char* ws = (char*)d_ws;
    size_t off = 0;
    auto take = [&](size_t bytes) -> char* {
        char* p = ws + off;
        off += (bytes + 255) & ~(size_t)255;
        return p;
    };
    float*  xw    = (float*)take((size_t)N * H0 * 4);
    float*  acc1  = (float*)take((size_t)N * H0 * 4);
    float*  hbuf  = (float*)take((size_t)N * H0 * 4);
    float*  xw2   = (float*)take((size_t)N * H1 * 4);
    float*  deg1  = (float*)take((size_t)N * 4);
    float*  dinv1 = (float*)take((size_t)N * 4);
    float*  deg2  = (float*)take((size_t)N * 4);
    float*  dinv2 = (float*)take((size_t)N * 4);
    float*  mw    = (float*)take((size_t)E * 4);
    __bf16* Bp1   = (__bf16*)take(16384 * 2);
    __bf16* Bp2   = (__bf16*)take(512 * 2);
    (void)ws_size; (void)n_in;

    hipMemsetAsync(deg1, 0, (size_t)N * 4, stream);
    hipMemsetAsync(deg2, 0, (size_t)N * 4, stream);
    hipMemsetAsync(acc1, 0, (size_t)N * H0 * 4, stream);
    hipMemsetAsync(out,  0, (size_t)out_size * 4, stream);

    ptdnet_pack_b1<<<64, 256, 0, stream>>>(W0, Bp1);
    ptdnet_pack_b2<<<2, 256, 0, stream>>>(W1, Bp2);

    int tiles = (N + 15) / 16;
    // Stage 1: xw = x @ W0 (WMMA bf16, dual accumulators per wave)
    ptdnet_gemm_xw<<<tiles, 32, 0, stream>>>(x, Bp1, xw, N);
    // Stage 1 aggregation
    ptdnet_degree_unit<<<(E + 255) / 256, 256, 0, stream>>>(col, deg1, E);
    ptdnet_rsqrt<<<(N + 255) / 256, 256, 0, stream>>>(deg1, dinv1, N);
    ptdnet_scatter1<<<(E + 127) / 128, 128, 0, stream>>>(xw, row, col, dinv1, acc1, E);
    ptdnet_finalize_h<<<((size_t)N * H0 + 255) / 256, 256, 0, stream>>>(acc1, xw, dinv1, b0, hbuf, N);
    // Stage 2: edge attention -> masked weights
    ptdnet_attention<<<(E + 255) / 256, 256, 0, stream>>>(hbuf, row, col, Wnb, bnb,
                                                          Wself, bself, Watt, batt, mw, E);
    // Stage 3: gated GCN
    ptdnet_degree_w<<<(E + 255) / 256, 256, 0, stream>>>(col, mw, deg2, E);
    ptdnet_rsqrt<<<(N + 255) / 256, 256, 0, stream>>>(deg2, dinv2, N);
    ptdnet_gemm_xw2<<<tiles, 32, 0, stream>>>(hbuf, Bp2, xw2, N);
    ptdnet_scatter2<<<(E + 255) / 256, 256, 0, stream>>>(xw2, row, col, dinv2, mw, out, E);
    ptdnet_finalize_out<<<((size_t)N * H1 + 255) / 256, 256, 0, stream>>>(xw2, dinv2, b1, out, N);
}
